// GCN_12197707120927
// MI455X (gfx1250) — compile-verified, matrix operand order checked by reading, not run on hardware
//
#include <hip/hip_runtime.h>

// ---------------------------------------------------------------------------
// Temporal GCN block on MI455X (gfx1250, wave32).
//  * heavy GEMMs on v_wmma_f32_16x16x32_bf16 (f32 accumulate)
//  * WMMA operands staged in LDS (xr_b fits entirely: 124KB of 320KB/WGP)
//  * xr_b -> LDS staging via the Tensor Data Mover when available
//  * top-k adjacency matmul replaced by sparse gather (~4 neighbors/row)
//  * gemm sized to 8 accumulator tiles/wave (64 VGPRs) to stay spill-free
// ---------------------------------------------------------------------------

typedef __attribute__((ext_vector_type(16))) __bf16 v16bf;
typedef __attribute__((ext_vector_type(8)))  float  v8f;
typedef unsigned int u32x4 __attribute__((ext_vector_type(4)));
typedef int          i32x8 __attribute__((ext_vector_type(8)));
typedef int          i32x4 __attribute__((ext_vector_type(4)));

#define BB   32
#define TT   243
#define JJ   17
#define CC   256
#define BJ   (BB * JJ)          // 544
#define MM   (BJ * TT)          // 132192 rows of xr
#define TP   256                // T padded to 16-tile multiple
#define KNEI 4

__device__ __forceinline__ unsigned short f2bf(float f) {
    unsigned int u = __float_as_uint(f);
    unsigned int r = u + 0x7FFFu + ((u >> 16) & 1u);   // round-to-nearest-even
    return (unsigned short)(r >> 16);
}

// One 16x32 bf16 operand tile (A or B share the per-lane K-run layout):
// lane%16 = row(A)/col(B), lane>=16 selects the second K-run.
// Elements 0..7 = K[half*8 .. +7], 8..15 = K[16+half*8 .. +7].
__device__ __forceinline__ v16bf load_tile(const unsigned short* base,
                                           int row, bool valid,
                                           int ld, int k0, int lane) {
    union { v16bf v; uint4 q[2]; } t;
    if (valid) {
        const unsigned short* p = base + (size_t)row * ld + k0 + ((lane >> 4) & 1) * 8;
        t.q[0] = *(const uint4*)(p);
        t.q[1] = *(const uint4*)(p + 16);
    } else {
        t.q[0] = make_uint4(0u, 0u, 0u, 0u);
        t.q[1] = make_uint4(0u, 0u, 0u, 0u);
    }
    return t.v;
}

__device__ __forceinline__ v8f wmma_bf16(v16bf a, v16bf b, v8f c) {
    return __builtin_amdgcn_wmma_f32_16x16x32_bf16(false, a, false, b,
                                                   (short)0, c, false, false);
}

// ---- kernel 1: x [B,T,J,C] -> xr [BJ,T,C] in bf16 -------------------------
__global__ void k_prep_x(const float* __restrict__ x, unsigned short* __restrict__ xr) {
    long blk = blockIdx.x;                 // bj*T + t
    int c  = threadIdx.x;
    int t  = (int)(blk % TT);
    int bj = (int)(blk / TT);
    int b  = bj / JJ, j = bj % JJ;
    float v = x[(((size_t)b * TT + t) * JJ + j) * CC + c];
    xr[(size_t)blk * CC + c] = f2bf(v);
}

__global__ void k_prep_w(const float* __restrict__ Wu, const float* __restrict__ Wv,
                         unsigned short* __restrict__ wu, unsigned short* __restrict__ wv) {
    long i = (long)blockIdx.x * blockDim.x + threadIdx.x;   // 0..65535
    wu[i] = f2bf(Wu[i]);
    wv[i] = f2bf(Wv[i]);
}

// ---- kernel 2: u = xr@Wu^T + bu,  v = xr@Wv^T + bv  (WMMA, LDS-staged) ----
// Block = 512 threads (16 waves) x 64 xr rows (4 M-tiles).  Per K-step the
// 32-wide slices of Wu, Wv (16KB each) and the 64x32 A slab (4KB) are staged
// in LDS.  Wave w: M-tile (w&3), matrix ((w>>2)&1), N-half (w>>3) -> 8
// accumulator tiles (64 VGPRs), 8 WMMAs per K-step off LDS, no spills.
__global__ void k_gemm_uv(const unsigned short* __restrict__ xr,
                          const unsigned short* __restrict__ wu,
                          const unsigned short* __restrict__ wv,
                          const float* __restrict__ bu, const float* __restrict__ bv,
                          float* __restrict__ u_out, float* __restrict__ v_out) {
    extern __shared__ unsigned short smem_us[];
    unsigned short* wuS = smem_us;             // [256][32]
    unsigned short* wvS = smem_us + 256 * 32;  // [256][32]
    unsigned short* aS  = smem_us + 512 * 32;  // [64][32]

    int tid  = threadIdx.x;
    int lane = tid & 31;
    int wave = tid >> 5;                    // 0..15
    int mt   = wave & 3;                    // M-tile within block
    int mat  = (wave >> 2) & 1;             // 0 = U, 1 = V
    int nh   = wave >> 3;                   // N-half: tiles nh*8 .. nh*8+7
    long row0 = (long)blockIdx.x * 64;

    v8f acc[8];
    #pragma unroll
    for (int i = 0; i < 8; ++i) acc[i] = v8f{};

    const unsigned short* wS = mat ? wvS : wuS;

    for (int k0 = 0; k0 < CC; k0 += 32) {
        // cooperative staging: 2304 uint4 = 36KB
        for (int q = tid; q < 2304; q += 512) {
            if (q < 1024) {                                // Wu slice
                int n = q >> 2, part = q & 3;
                *(uint4*)(wuS + n * 32 + part * 8) =
                    *(const uint4*)(wu + (size_t)n * CC + k0 + part * 8);
            } else if (q < 2048) {                         // Wv slice
                int n = (q - 1024) >> 2, part = q & 3;
                *(uint4*)(wvS + n * 32 + part * 8) =
                    *(const uint4*)(wv + (size_t)n * CC + k0 + part * 8);
            } else {                                       // A slab
                int r = (q - 2048) >> 2, part = q & 3;
                long grow = row0 + r;
                uint4 val = make_uint4(0u, 0u, 0u, 0u);
                if (grow < MM)
                    val = *(const uint4*)(xr + (size_t)grow * CC + k0 + part * 8);
                *(uint4*)(aS + r * 32 + part * 8) = val;
            }
        }
        __syncthreads();

        v16bf a = load_tile(aS, mt * 16 + (lane & 15), true, 32, 0, lane);
        #pragma unroll
        for (int i = 0; i < 8; ++i) {
            int nt = nh * 8 + i;
            v16bf bt = load_tile(wS, nt * 16 + (lane & 15), true, 32, 0, lane);
            acc[i] = wmma_bf16(a, bt, acc[i]);
        }
        __syncthreads();
    }

    int half = lane >> 4;
    int nc   = lane & 15;
    const float* bias = mat ? bv : bu;
    float* outp = mat ? v_out : u_out;
    #pragma unroll
    for (int i = 0; i < 8; ++i) {
        int n = (nh * 8 + i) * 16 + nc;
        float add = bias[n];
        #pragma unroll
        for (int e = 0; e < 8; ++e) {
            long m = row0 + mt * 16 + e + 8 * half;
            if (m < MM) outp[(size_t)m * CC + n] = acc[i][e] + add;
        }
    }
}

// ---- kernel 3: sim[b] = xr_b @ xr_b^T (243x243, padded), xr_b in LDS ------
__global__ void k_sim(const unsigned short* __restrict__ xr, float* __restrict__ sim) {
    extern __shared__ unsigned short xs[];              // [243][256] bf16 = 124KB
    int b = blockIdx.x;
    const unsigned short* xb = xr + (size_t)b * TT * CC;
    float* sb = sim + (size_t)b * TT * TP;
    int lane = threadIdx.x & 31;
    int wave = threadIdx.x >> 5;

#if defined(__has_builtin) && __has_builtin(__builtin_amdgcn_tensor_load_to_lds)
    // Tensor Data Mover: DMA the whole 243x256 bf16 tile into LDS.
    if (wave == 0) {
        unsigned long long ga = (unsigned long long)xb;
        unsigned lds_addr = (unsigned)(unsigned long long)(void*)xs;
        u32x4 g0 = { 1u,                                       // count=1 (user D#)
                     lds_addr,                                 // lds_addr [63:32]
                     (unsigned)(ga & 0xFFFFFFFFu),             // global_addr lo
                     (unsigned)((ga >> 32) & 0x1FFFFFFu) | (2u << 30) }; // type=2
        i32x8 g1 = { (int)(1u << 16),                 // data_size=1 -> 2 bytes
                     (int)((unsigned)(CC & 0xFFFF) << 16),  // tensor_dim0 lo16
                     (int)((unsigned)(TT & 0xFFFF) << 16),  // tensor_dim1 lo16
                     (int)((unsigned)CC << 16),       // tile_dim0 = 256
                     TT,                              // tile_dim1 = 243 (tile_dim2=0)
                     CC,                              // tensor_dim0_stride lo32
                     0, 0 };
        i32x4 gz = { 0, 0, 0, 0 };
#if __clang_major__ >= 23
        i32x8 gz8 = { 0, 0, 0, 0, 0, 0, 0, 0 };
        __builtin_amdgcn_tensor_load_to_lds(g0, g1, gz, gz, gz8, 0);
#else
        __builtin_amdgcn_tensor_load_to_lds(g0, g1, gz, gz, 0);
#endif
        __builtin_amdgcn_s_wait_tensorcnt(0);
    }
    __syncthreads();
#else
    for (int q = threadIdx.x; q < (TT * CC) / 8; q += 256)   // 7776 x uint4
        *(uint4*)(xs + q * 8) = *(const uint4*)(xb + q * 8);
    __syncthreads();
#endif

    for (int tt = wave; tt < 256; tt += 8) {        // 16x16 tile grid over TPxTP
        int ti = tt >> 4, tj = tt & 15;
        v8f acc = v8f{};
        for (int k0 = 0; k0 < CC; k0 += 32) {
            int ra = ti * 16 + (lane & 15);
            int rb = tj * 16 + (lane & 15);
            v16bf a  = load_tile(xs, ra, ra < TT, CC, k0, lane);
            v16bf bm = load_tile(xs, rb, rb < TT, CC, k0, lane); // B[k][n]=xr[n][k]
            acc = wmma_bf16(a, bm, acc);
        }
        int half = lane >> 4;
        int n = tj * 16 + (lane & 15);
        #pragma unroll
        for (int e = 0; e < 8; ++e) {
            int m = ti * 16 + e + 8 * half;
            if (m < TT) sb[(size_t)m * TP + n] = acc[e];
        }
    }
}

// ---- kernel 4: per-row top-4 threshold + degree^{-1/2} (wave per row) -----
__global__ void k_topk(const float* __restrict__ sim,
                       float* __restrict__ thr, float* __restrict__ dinv) {
    int lane = threadIdx.x & 31;
    int wave = threadIdx.x >> 5;
    long row = (long)blockIdx.x * 8 + wave;          // 16524*8 == MM exactly
    const float* sr = sim + row * TP;

    const float NEG = -__builtin_inff();
    float vals[8];
    bool  act[8];
    #pragma unroll
    for (int s = 0; s < 8; ++s) {
        int idx = lane + s * 32;
        bool ok = idx < TT;
        vals[s] = ok ? sr[idx] : NEG;
        act[s]  = ok;
    }

    float g = NEG;
    for (int it = 0; it < KNEI; ++it) {              // duplicate-aware top-k
        float lm = NEG;
        #pragma unroll
        for (int s = 0; s < 8; ++s) if (act[s] && vals[s] > lm) lm = vals[s];
        float m = lm;
        #pragma unroll
        for (int o = 16; o > 0; o >>= 1) m = fmaxf(m, __shfl_xor(m, o, 32));
        g = m;
        unsigned long long bal = __ballot(lm == m);
        int first = __ffsll(bal) - 1;
        if (lane == first) {
            bool done = false;
            #pragma unroll
            for (int s = 0; s < 8; ++s)
                if (!done && act[s] && vals[s] == m) { act[s] = false; done = true; }
        }
    }

    int cnt = 0;
    #pragma unroll
    for (int s = 0; s < 8; ++s)
        if ((lane + s * 32) < TT && vals[s] >= g) cnt++;
    #pragma unroll
    for (int o = 16; o > 0; o >>= 1) cnt += __shfl_xor(cnt, o, 32);

    if (lane == 0) {
        thr[row]  = g;
        dinv[row] = rsqrtf((float)cnt);
    }
}

// ---- kernel 5: sparse aggregation h = D^-1/2 A D^-1/2 v + u (in place) ----
__global__ void k_agg(const float* __restrict__ sim, const float* __restrict__ thr,
                      const float* __restrict__ dinv, const float* __restrict__ v,
                      float* __restrict__ h) {
    long row = blockIdx.x;                 // b*TT + n
    int  b   = (int)(row / TT);
    int  c   = threadIdx.x;

    __shared__ float srow[TP];
    __shared__ float sd[TP];
    srow[c] = sim[row * TP + c];
    sd[c]   = (c < TT) ? dinv[(size_t)b * TT + c] : 0.0f;
    __syncthreads();

    float tn = thr[row];
    float dn = dinv[row];
    const float* vb = v + (size_t)b * TT * CC;

    float acc = 0.0f;
    for (int m = 0; m < TT; ++m) {
        if (srow[m] >= tn)                 // uniform branch: ~4 taken per row
            acc += sd[m] * vb[(size_t)m * CC + c];
    }
    size_t o = row * (size_t)CC + c;
    h[o] = dn * acc + h[o];                // h buffer was pre-loaded with u
}

// ---- kernel 6: BatchNorm1d(T) stats over (BJ, C) -> fused scale/shift -----
__global__ void k_bn(const float* __restrict__ h, const float* __restrict__ gamma,
                     const float* __restrict__ beta,
                     float* __restrict__ scale, float* __restrict__ shift) {
    int t = blockIdx.x, c = threadIdx.x;
    float s = 0.0f, ss = 0.0f;
    for (int bj = 0; bj < BJ; ++bj) {
        float x = h[((size_t)bj * TT + t) * CC + c];
        s += x; ss += x * x;
    }
    __shared__ float rs[256], rq[256];
    rs[c] = s; rq[c] = ss;
    __syncthreads();
    for (int st = 128; st > 0; st >>= 1) {
        if (c < st) { rs[c] += rs[c + st]; rq[c] += rq[c + st]; }
        __syncthreads();
    }
    if (c == 0) {
        const float N = (float)BJ * (float)CC;
        float mean = rs[0] / N;
        float var  = rq[0] / N - mean * mean;
        float sc   = gamma[t] * rsqrtf(var + 1e-5f);
        scale[t] = sc;
        shift[t] = beta[t] - mean * sc;
    }
}

// ---- kernel 7: out = relu(x + bn(h)), back to [B,T,J,C] -------------------
__global__ void k_final(const float* __restrict__ x, const float* __restrict__ h,
                        const float* __restrict__ scale, const float* __restrict__ shift,
                        float* __restrict__ out) {
    long blk = blockIdx.x;                 // (b*TT + t)*JJ + j
    int  c   = threadIdx.x;
    int  j   = (int)(blk % JJ);
    long bt  = blk / JJ;
    int  t   = (int)(bt % TT);
    int  b   = (int)(bt / TT);
    int  bj  = b * JJ + j;

    size_t xo = (size_t)blk * CC + c;
    size_t ho = ((size_t)bj * TT + t) * (size_t)CC + c;
    float val = x[xo] + h[ho] * scale[t] + shift[t];
    out[xo] = fmaxf(val, 0.0f);
}

// ---------------------------------------------------------------------------
extern "C" void kernel_launch(void* const* d_in, const int* in_sizes, int n_in,
                              void* d_out, int out_size, void* d_ws, size_t ws_size,
                              hipStream_t stream) {
    (void)in_sizes; (void)n_in; (void)out_size; (void)ws_size;
    const float* x     = (const float*)d_in[0];
    const float* W_u   = (const float*)d_in[1];
    const float* b_u   = (const float*)d_in[2];
    const float* W_v   = (const float*)d_in[3];
    const float* b_v   = (const float*)d_in[4];
    const float* gamma = (const float*)d_in[5];
    const float* beta  = (const float*)d_in[6];
    float* out = (float*)d_out;

    char* ws = (char*)d_ws;
    size_t off = 0;
    auto alloc = [&](size_t bytes) {
        size_t o = off;
        off = (off + bytes + 255) & ~(size_t)255;
        return o;
    };
    unsigned short* xr  = (unsigned short*)(ws + alloc((size_t)MM * CC * 2));
    unsigned short* wu  = (unsigned short*)(ws + alloc((size_t)CC * CC * 2));
    unsigned short* wv  = (unsigned short*)(ws + alloc((size_t)CC * CC * 2));
    float* vbuf  = (float*)(ws + alloc((size_t)MM * CC * 4));
    float* hbuf  = (float*)(ws + alloc((size_t)MM * CC * 4));   // u, then h in place
    float* sbuf  = (float*)(ws + alloc((size_t)MM * TP * 4));
    float* thr   = (float*)(ws + alloc((size_t)MM * 4));
    float* dinv  = (float*)(ws + alloc((size_t)MM * 4));
    float* scale = (float*)(ws + alloc((size_t)TT * 4));
    float* shift = (float*)(ws + alloc((size_t)TT * 4));

    const size_t simLds  = (size_t)TT * CC * 2;         // 124416 B
    const size_t gemmLds = (size_t)(512 + 64) * 32 * 2; // 36864 B

    k_prep_x <<<MM, 256, 0, stream>>>(x, xr);
    k_prep_w <<<(CC * CC) / 256, 256, 0, stream>>>(W_u, W_v, wu, wv);
    k_gemm_uv<<<(MM + 63) / 64, 512, gemmLds, stream>>>(xr, wu, wv, b_u, b_v, hbuf, vbuf);
    k_sim    <<<BJ, 256, simLds, stream>>>(xr, sbuf);
    k_topk   <<<MM / 8, 256, 0, stream>>>(sbuf, thr, dinv);
    k_agg    <<<MM, 256, 0, stream>>>(sbuf, thr, dinv, vbuf, hbuf);
    k_bn     <<<TT, 256, 0, stream>>>(hbuf, gamma, beta, scale, shift);
    k_final  <<<MM, 256, 0, stream>>>(x, hbuf, scale, shift, out);
}
